// MultiHeadAttention_45200235823485
// MI455X (gfx1250) — matrix operand order repaired and checked by reading, not compiled, not run on hardware
//
#include <hip/hip_runtime.h>
#include <math.h>

#define DM   768
#define SEQ  2048
#define BATCH 4
#define NH   12
#define DK   64
#define ROWS (BATCH * SEQ)        // 8192
#define NPROJ (3 * DM)            // 2304

typedef __attribute__((ext_vector_type(16))) __bf16          v16bf;
typedef __attribute__((ext_vector_type(8)))  float           v8f;
typedef unsigned short u16;
typedef __attribute__((ext_vector_type(8)))  unsigned short  v8u;
typedef __attribute__((ext_vector_type(16))) unsigned short  v16u;

union V16U { v16u v; v8u h[2]; };

__device__ __forceinline__ u16 f2bf(float f) {
    unsigned u = __builtin_bit_cast(unsigned, f);
    u += 0x7FFFu + ((u >> 16) & 1u);            // round-to-nearest-even
    return (u16)(u >> 16);
}

// A-matrix 16x32 bf16 fragment: per-lane two contiguous 8-element (16B) chunks:
// features [base + hi*16 .. +7] and [base + hi*16 + 8 .. +7]; caller bakes hi*16 into p.
__device__ __forceinline__ v16u load_afrag(const u16* p) {
    V16U t;
    t.h[0] = *(const v8u*)p;
    t.h[1] = *(const v8u*)(p + 8);
    return t.v;
}

__device__ __forceinline__ v8f wmma_bf(v16u a, v16u b, v8f c) {
    return __builtin_amdgcn_wmma_f32_16x16x32_bf16(
        false, __builtin_bit_cast(v16bf, a),
        false, __builtin_bit_cast(v16bf, b),
        (short)0, c, false, false);
}

template <int MASK>
__device__ __forceinline__ float xor_f32(float v) {
    int x = __builtin_bit_cast(int, v);
    x = __builtin_amdgcn_ds_swizzle(x, (MASK << 10) | 0x1F);  // xor-mask swizzle
    return __builtin_bit_cast(float, x);
}

// ---------------------------------------------------------------- convert
__global__ void __launch_bounds__(256) cvt_kernel(const float* __restrict__ in,
                                                  u16* __restrict__ out, int n) {
    int i = blockIdx.x * 256 + threadIdx.x;
    if (i < n) out[i] = f2bf(in[i]);
}

// ---------------------------------------------------------------- fused QKV GEMM
// out[s, e] = sum_d x[s,d] * W[e,d] + b[e], e spans {Q,K,V} x 768.
// Q,K -> [BH, S, 64] bf16 ; V -> transposed [BH, 64, S] bf16.
__global__ void __launch_bounds__(256) qkv_kernel(
    const u16* __restrict__ xbf,
    const u16* __restrict__ wq, const u16* __restrict__ wk, const u16* __restrict__ wv,
    const float* __restrict__ bq, const float* __restrict__ bk, const float* __restrict__ bv,
    u16* __restrict__ Qb, u16* __restrict__ Kb, u16* __restrict__ Vt) {
    const int lane = threadIdx.x & 31;
    const int wave = threadIdx.x >> 5;
    const int lo = lane & 15, hi = lane >> 4;
    const int wid = blockIdx.x * 8 + wave;        // 0..9215
    const int tm = wid / (NPROJ / 64);            // 0..255
    const int tn = wid % (NPROJ / 64);            // 0..35
    const int m0 = tm * 32;
    const int n0 = tn * 64;
    const int which = n0 / DM;                    // 0=Q 1=K 2=V
    const int nloc = n0 % DM;
    const u16* W = (which == 0) ? wq : (which == 1 ? wk : wv);
    const float* bias = (which == 0) ? bq : (which == 1 ? bk : bv);

    v8f acc[2][4] = {};
    const u16* ar0 = xbf + (size_t)(m0 + lo) * DM + hi * 16;
    const u16* ar1 = ar0 + (size_t)16 * DM;
    const u16* bc[4];
#pragma unroll
    for (int ni = 0; ni < 4; ++ni)
        bc[ni] = W + (size_t)(nloc + ni * 16 + lo) * DM + hi * 16;

    for (int kk = 0; kk < DM; kk += 32) {
        v16u a0 = load_afrag(ar0 + kk);
        v16u a1 = load_afrag(ar1 + kk);
#pragma unroll
        for (int ni = 0; ni < 4; ++ni) {
            v16u bfr = *(const v16u*)(bc[ni] + kk);
            acc[0][ni] = wmma_bf(a0, bfr, acc[0][ni]);
            acc[1][ni] = wmma_bf(a1, bfr, acc[1][ni]);
        }
    }

#pragma unroll
    for (int ni = 0; ni < 4; ++ni) {
        const int n = nloc + ni * 16 + lo;
        const float bv_ = bias[n];
        const int h = n >> 6, d = n & 63;
#pragma unroll
        for (int mi = 0; mi < 2; ++mi) {
            const int mbase = m0 + mi * 16 + hi * 8;    // 8 consecutive rows
            const int b = mbase >> 11;
            const int s = mbase & 2047;
            if (which == 2) {                           // V transposed: contiguous in s
                v8u pk;
#pragma unroll
                for (int r = 0; r < 8; ++r) pk[r] = f2bf(acc[mi][ni][r] + bv_);
                *(v8u*)(Vt + (((size_t)(b * NH + h) * DK + d) * SEQ + s)) = pk;
            } else {
                u16* dst = (which == 0 ? Qb : Kb) +
                           (((size_t)(b * NH + h) * SEQ + s) * DK + d);
#pragma unroll
                for (int r = 0; r < 8; ++r) dst[(size_t)r * DK] = f2bf(acc[mi][ni][r] + bv_);
            }
        }
    }
}

// ---------------------------------------------------------------- flash attention
// One wave per (bh, 16-query tile). 32-key blocks, online softmax, exp2 path.
__global__ void __launch_bounds__(256) flash_kernel(
    const u16* __restrict__ Qb, const u16* __restrict__ Kb,
    const u16* __restrict__ Vt, u16* __restrict__ ctx) {
    __shared__ u16 Plds[8][16 * 32];
    const int lane = threadIdx.x & 31;
    const int wave = threadIdx.x >> 5;
    const int lo = lane & 15, hi = lane >> 4;
    const int wid = blockIdx.x * 8 + wave;        // 0..6143
    const int qt = wid & 127;
    const int bh = wid >> 7;
    const int q0 = qt << 4;

    const u16* Qp = Qb + (size_t)bh * SEQ * DK;
    const u16* Kp = Kb + (size_t)bh * SEQ * DK;
    const u16* Vp = Vt + (size_t)bh * DK * SEQ;

    v16u qa[2];
    {
        const u16* qr = Qp + (size_t)(q0 + lo) * DK + hi * 16;
        qa[0] = load_afrag(qr);          // d = 0..31
        qa[1] = load_afrag(qr + 32);     // d = 32..63
    }

    float mst[8], lst[8];
#pragma unroll
    for (int r = 0; r < 8; ++r) { mst[r] = -1e30f; lst[r] = 0.f; }
    v8f o[4] = {};

    const float sc = 0.125f * 1.4426950408889634f;   // 1/sqrt(64) * log2(e)
    const int nkb = (q0 + 47) >> 5;                  // causal: keys 0 .. q0+15

    for (int j = 0; j < nkb; ++j) {
        const int kb = j << 5;
        v8f s0 = {}, s1 = {};
#pragma unroll
        for (int ds = 0; ds < 2; ++ds) {
            v16u k0 = *(const v16u*)(Kp + (size_t)(kb + lo) * DK + ds * 32 + hi * 16);
            v16u k1 = *(const v16u*)(Kp + (size_t)(kb + 16 + lo) * DK + ds * 32 + hi * 16);
            s0 = wmma_bf(qa[ds], k0, s0);
            s1 = wmma_bf(qa[ds], k1, s1);
        }
        const bool msk = (kb + 31 > q0);
        float corr[8];
#pragma unroll
        for (int r = 0; r < 8; ++r) {
            float t0 = s0[r] * sc;
            float t1 = s1[r] * sc;
            if (msk) {
                int q = q0 + r + hi * 8;             // C-layout row for this VGPR
                if (kb + lo > q)      t0 = -1e30f;   // key index = kb + lo
                if (kb + 16 + lo > q) t1 = -1e30f;
            }
            float mx = fmaxf(t0, t1);
            mx = fmaxf(mx, xor_f32<1>(mx));
            mx = fmaxf(mx, xor_f32<2>(mx));
            mx = fmaxf(mx, xor_f32<4>(mx));
            mx = fmaxf(mx, xor_f32<8>(mx));
            float mnew = fmaxf(mst[r], mx);
            float c  = exp2f(mst[r] - mnew);
            float p0 = exp2f(t0 - mnew);
            float p1 = exp2f(t1 - mnew);
            mst[r] = mnew;
            corr[r] = c;
            float ps = p0 + p1;
            ps += xor_f32<1>(ps);
            ps += xor_f32<2>(ps);
            ps += xor_f32<4>(ps);
            ps += xor_f32<8>(ps);
            lst[r] = lst[r] * c + ps;
            const int row = r + hi * 8;
            Plds[wave][row * 32 + lo]      = f2bf(p0);
            Plds[wave][row * 32 + 16 + lo] = f2bf(p1);
        }
#pragma unroll
        for (int nt = 0; nt < 4; ++nt)
#pragma unroll
            for (int r = 0; r < 8; ++r) o[nt][r] *= corr[r];

        // P (16x32) as A fragment via LDS (DS ops are in-order within the wave)
        v16u pa = load_afrag(&Plds[wave][lo * 32 + hi * 16]);
#pragma unroll
        for (int nt = 0; nt < 4; ++nt) {
            v16u vb = *(const v16u*)(Vp + (size_t)(nt * 16 + lo) * SEQ + kb + hi * 16);
            o[nt] = wmma_bf(pa, vb, o[nt]);
        }
    }

    const int b = bh / NH, h = bh % NH;
    u16* crow = ctx + (size_t)(b * SEQ + q0 + hi * 8) * DM + h * DK;
#pragma unroll
    for (int r = 0; r < 8; ++r) {
        const float rinv = 1.0f / lst[r];
#pragma unroll
        for (int nt = 0; nt < 4; ++nt)
            crow[(size_t)r * DM + nt * 16 + lo] = f2bf(o[nt][r] * rinv);
    }
}

// ---------------------------------------------------------------- output projection
__global__ void __launch_bounds__(256) oproj_kernel(
    const u16* __restrict__ ctx, const u16* __restrict__ wo,
    const float* __restrict__ bo, float* __restrict__ out) {
    const int lane = threadIdx.x & 31;
    const int wave = threadIdx.x >> 5;
    const int lo = lane & 15, hi = lane >> 4;
    const int wid = blockIdx.x * 8 + wave;     // 0..3071
    const int tm = wid / (DM / 64);
    const int tn = wid % (DM / 64);
    const int m0 = tm * 32;
    const int n0 = tn * 64;

    v8f acc[2][4] = {};
    const u16* ar0 = ctx + (size_t)(m0 + lo) * DM + hi * 16;
    const u16* ar1 = ar0 + (size_t)16 * DM;
    const u16* bc[4];
#pragma unroll
    for (int ni = 0; ni < 4; ++ni)
        bc[ni] = wo + (size_t)(n0 + ni * 16 + lo) * DM + hi * 16;

    for (int kk = 0; kk < DM; kk += 32) {
        v16u a0 = load_afrag(ar0 + kk);
        v16u a1 = load_afrag(ar1 + kk);
#pragma unroll
        for (int ni = 0; ni < 4; ++ni) {
            v16u bfr = *(const v16u*)(bc[ni] + kk);
            acc[0][ni] = wmma_bf(a0, bfr, acc[0][ni]);
            acc[1][ni] = wmma_bf(a1, bfr, acc[1][ni]);
        }
    }

#pragma unroll
    for (int ni = 0; ni < 4; ++ni) {
        const int n = n0 + ni * 16 + lo;
        const float bv_ = bo[n];
#pragma unroll
        for (int mi = 0; mi < 2; ++mi) {
            float* dst = out + (size_t)(m0 + mi * 16 + hi * 8) * DM + n;
#pragma unroll
            for (int r = 0; r < 8; ++r) dst[(size_t)r * DM] = acc[mi][ni][r] + bv_;
        }
    }
}

// ---------------------------------------------------------------- launch
extern "C" void kernel_launch(void* const* d_in, const int* in_sizes, int n_in,
                              void* d_out, int out_size, void* d_ws, size_t ws_size,
                              hipStream_t stream) {
    const float* x  = (const float*)d_in[0];
    const float* wq = (const float*)d_in[1];
    const float* bq = (const float*)d_in[2];
    const float* wk = (const float*)d_in[3];
    const float* bk = (const float*)d_in[4];
    const float* wv = (const float*)d_in[5];
    const float* bv = (const float*)d_in[6];
    const float* wo = (const float*)d_in[7];
    const float* bo = (const float*)d_in[8];
    float* out = (float*)d_out;

    const size_t NX = (size_t)ROWS * DM;   // 6,291,456
    const size_t NW = (size_t)DM * DM;     //   589,824
    u16* p = (u16*)d_ws;
    u16* xbf = p; p += NX;
    u16* wqb = p; p += NW;
    u16* wkb = p; p += NW;
    u16* wvb = p; p += NW;
    u16* wob = p; p += NW;
    u16* Qb  = p; p += NX;
    u16* Kb  = p; p += NX;
    u16* Vtb = p; p += NX;
    u16* ctx = p;                          // total ~67.6 MB of ws

    cvt_kernel<<<(int)((NX + 255) / 256), 256, 0, stream>>>(x,  xbf, (int)NX);
    cvt_kernel<<<(int)((NW + 255) / 256), 256, 0, stream>>>(wq, wqb, (int)NW);
    cvt_kernel<<<(int)((NW + 255) / 256), 256, 0, stream>>>(wk, wkb, (int)NW);
    cvt_kernel<<<(int)((NW + 255) / 256), 256, 0, stream>>>(wv, wvb, (int)NW);
    cvt_kernel<<<(int)((NW + 255) / 256), 256, 0, stream>>>(wo, wob, (int)NW);

    qkv_kernel<<<1152, 256, 0, stream>>>(xbf, wqb, wkb, wvb, bq, bk, bv, Qb, Kb, Vtb);
    flash_kernel<<<768, 256, 0, stream>>>(Qb, Kb, Vtb, ctx);
    oproj_kernel<<<384, 256, 0, stream>>>(ctx, wob, bo, out);
}
